// Net_60696477827134
// MI455X (gfx1250) — compile-verified, matrix operand order checked by reading, not run on hardware
//
#include <hip/hip_runtime.h>
#include <math.h>

// ---------------------------------------------------------------------------
// CDNA5 / gfx1250 MoE MLP forward.  Weight-streaming bound (~530MB fp32):
// b128 global loads -> LDS (fp32->bf16 convert once) -> v_wmma_f32_16x16x32_bf16
// with fp32 accumulate.  Double-buffered LDS, 1 barrier / K-chunk.  wave32.
// ---------------------------------------------------------------------------

typedef __attribute__((ext_vector_type(16))) __bf16 v16bf;
typedef __attribute__((ext_vector_type(8)))  float  v8f;

#define BM 128
#define BN 128
#define BK 32
#define LDSS 34   // ushort stride per tile row (68B -> 17-dword stride, no bank conflicts)

__device__ __forceinline__ unsigned short f2bf(float f) {
    unsigned u = __float_as_uint(f);
    u += 0x7FFFu + ((u >> 16) & 1u); // round-to-nearest-even
    return (unsigned short)(u >> 16);
}

struct GemmP {
    const float*          Bw;      // weights [K, N] row-major
    const float*          bias;    // [N]
    const float*          A32;     // fp32 A [BM, lda] (fc0) or null
    const unsigned short* A16;     // bf16 A [BM, lda] or null
    const int*            gather;  // per-row expert index (fco) or null
    long                  gstride; // elements per gather index
    float*                out32;   // fp32 out [BM, N] or null
    unsigned short*       out16;   // bf16 out [BM, N] or null
    int                   Kreal;   // true K (may be non-multiple of 32)
    int                   lda;     // A row stride (elements)
    int                   N;       // true N == B/out row stride
    long strideB_e, strideBias_e, strideA_e, strideO_e; // per blockIdx.y
};

template <bool AF32>
__global__ __launch_bounds__(256) void gemm_wmma(GemmP p) {
    __shared__ unsigned short As[2][BM * LDSS]; // A tile: row m, 32 bf16 (K pairs)
    __shared__ unsigned short Bs[2][BN * LDSS]; // B tile: row = column n, 32 bf16 (K pairs)
    __shared__ int idxLds[BM];

    const int e = blockIdx.y;
    const float* Bw = p.Bw + (long)e * p.strideB_e;
    const float* bias = p.bias + (long)e * p.strideBias_e;
    const unsigned short* A16 = p.A16 ? p.A16 + (long)e * p.strideA_e : nullptr;
    unsigned short* out16 = p.out16 ? p.out16 + (long)e * p.strideO_e : nullptr;

    const int tid   = threadIdx.x;
    const int wave  = tid >> 5;          // 0..7 -> 16-column slice
    const int lane  = tid & 31;
    const int bn0   = blockIdx.x * BN;
    const int nloc  = wave * 16 + (lane & 15);
    const int nc    = bn0 + nloc;
    const int ncl   = (nc < p.N) ? nc : (p.N - 1); // clamped for loads
    const int khalf = (lane >> 4) * 8;   // K offset for high half-lanes

    if (tid < BM) idxLds[tid] = (!AF32 && p.gather) ? p.gather[tid] : 0;
    __syncthreads();

    // ---- staging registers (next chunk, global -> reg) ----
    float4 br0[2], br1[2];   // B: 2 work items, K-pair of float4 columns
    float2 a32r[8];          // A fp32: 8 work items (m, K-pair)
    uint4  a16r[2];          // A bf16: 2 work items (m, 8-K group)

    auto stage_global = [&](int k0) {
        // B chunk [32 x 128] fp32, b128 coalesced
#pragma unroll
        for (int w = 0; w < 2; ++w) {
            int id = tid + 256 * w;       // 0..511
            int kp = id >> 5, ng = id & 31;
            int kb = k0 + 2 * kp;
            int n  = bn0 + 4 * ng;
            float4 r0 = {0.f, 0.f, 0.f, 0.f}, r1 = {0.f, 0.f, 0.f, 0.f};
            if (kb < p.Kreal) {
                if (n + 3 < p.N) {
                    r0 = *(const float4*)(Bw + (long)kb * p.N + n);
                    if (kb + 1 < p.Kreal)
                        r1 = *(const float4*)(Bw + (long)(kb + 1) * p.N + n);
                } else { // ragged N (fco): clamped scalar loads
#pragma unroll
                    for (int j = 0; j < 4; ++j) {
                        int nn = (n + j < p.N) ? n + j : p.N - 1;
                        ((float*)&r0)[j] = Bw[(long)kb * p.N + nn];
                        if (kb + 1 < p.Kreal)
                            ((float*)&r1)[j] = Bw[(long)(kb + 1) * p.N + nn];
                    }
                }
            }
            br0[w] = r0; br1[w] = r1;
        }
        // A chunk [128 x 32]
        if (AF32) {
#pragma unroll
            for (int w = 0; w < 8; ++w) {
                int id = tid + 256 * w;   // 0..2047
                int m = id >> 4, kp = id & 15;
                int kb = k0 + 2 * kp;
                float2 r = {0.f, 0.f};
                if (kb < p.Kreal)         // Kreal even -> pair fully in-bounds
                    r = *(const float2*)(p.A32 + (long)m * p.lda + kb);
                a32r[w] = r;
            }
        } else {
#pragma unroll
            for (int w = 0; w < 2; ++w) {
                int id = tid + 256 * w;   // 0..511
                int m = id >> 2, g = id & 3;
                long goff = p.gather ? (long)idxLds[m] * p.gstride : 0;
                a16r[w] = *(const uint4*)(A16 + goff + (long)m * p.lda + k0 + 8 * g);
            }
        }
    };

    auto stage_lds = [&](int buf) {
        unsigned short* Asb = &As[buf][0];
        unsigned short* Bsb = &Bs[buf][0];
#pragma unroll
        for (int w = 0; w < 2; ++w) {
            int id = tid + 256 * w;
            int kp = id >> 5, ng = id & 31;
#pragma unroll
            for (int j = 0; j < 4; ++j) {
                unsigned u = (unsigned)f2bf(((const float*)&br0[w])[j]) |
                             ((unsigned)f2bf(((const float*)&br1[w])[j]) << 16);
                *(unsigned*)&Bsb[(4 * ng + j) * LDSS + 2 * kp] = u;
            }
        }
        if (AF32) {
#pragma unroll
            for (int w = 0; w < 8; ++w) {
                int id = tid + 256 * w;
                int m = id >> 4, kp = id & 15;
                unsigned u = (unsigned)f2bf(a32r[w].x) | ((unsigned)f2bf(a32r[w].y) << 16);
                *(unsigned*)&Asb[m * LDSS + 2 * kp] = u;
            }
        } else {
#pragma unroll
            for (int w = 0; w < 2; ++w) {
                int id = tid + 256 * w;
                int m = id >> 2, g = id & 3;
                unsigned* dst = (unsigned*)&Asb[m * LDSS + 8 * g];
                const unsigned* s = (const unsigned*)&a16r[w];
                dst[0] = s[0]; dst[1] = s[1]; dst[2] = s[2]; dst[3] = s[3];
            }
        }
    };

    v8f acc[8];
    const v8f vzero = {0.f, 0.f, 0.f, 0.f, 0.f, 0.f, 0.f, 0.f};
#pragma unroll
    for (int t = 0; t < 8; ++t) acc[t] = vzero;

    auto compute = [&](int buf) {
        const unsigned short* Asb = &As[buf][0];
        const unsigned short* Bsb = &Bs[buf][0];
        union { v16bf v; unsigned u[8]; } bf;
#pragma unroll
        for (int v = 0; v < 8; ++v) {
            int kb = ((v & 4) ? 16 : 0) + khalf + 2 * (v & 3);
            bf.u[v] = *(const unsigned*)&Bsb[nloc * LDSS + kb];
        }
#pragma unroll
        for (int t = 0; t < 8; ++t) {
            union { v16bf v; unsigned u[8]; } af;
            const int m = t * 16 + (lane & 15);
#pragma unroll
            for (int v = 0; v < 8; ++v) {
                int kb = ((v & 4) ? 16 : 0) + khalf + 2 * (v & 3);
                af.u[v] = *(const unsigned*)&Asb[m * LDSS + kb];
            }
            acc[t] = __builtin_amdgcn_wmma_f32_16x16x32_bf16(
                false, af.v, false, bf.v, (short)0, acc[t], false, false);
        }
    };

    // ---- software pipeline: global(c+1) || wmma(c), then convert+store, 1 sync ----
    const int nch = (p.Kreal + BK - 1) / BK;
    stage_global(0);
    stage_lds(0);
    __syncthreads();
    for (int c = 0; c < nch; ++c) {
        if (c + 1 < nch) stage_global((c + 1) * BK);
        compute(c & 1);
        if (c + 1 < nch) stage_lds((c + 1) & 1);
        __syncthreads();
    }

    // ---- epilogue: bias + relu, store fp32 and/or bf16 ----
    const float bv = bias[ncl];
#pragma unroll
    for (int t = 0; t < 8; ++t) {
#pragma unroll
        for (int v = 0; v < 8; ++v) {
            int m = t * 16 + v + ((lane >> 4) * 8);
            float r = acc[t][v] + bv;
            r = r > 0.f ? r : 0.f;
            if (nc < p.N) {
                long o = (long)m * p.N + nc;
                if (p.out32) p.out32[o] = r;
                if (out16)   out16[o]   = f2bf(r);
            }
        }
    }
}

// ---------------------------------------------------------------------------
// selector softmax + argmax routing + sigmoid chain + synthetic loss
// ---------------------------------------------------------------------------
__global__ __launch_bounds__(1024) void selector_kernel(
    const float* __restrict__ h, const float* __restrict__ sel_w,
    const float* __restrict__ sel_b, const float* __restrict__ sg_w,
    const float* __restrict__ sg_b, const float* __restrict__ sgo_w,
    const float* __restrict__ sgo_b, const float* __restrict__ sel_loss,
    int* __restrict__ idx_out, float* __restrict__ d_out) {
    __shared__ float synL[128];
    const int tid = threadIdx.x, wave = tid >> 5, lane = tid & 31;
    for (int b = wave; b < 128; b += 32) {
        float s0 = 0.f, s1 = 0.f, s2 = 0.f;
        for (int k = lane; k < 8192; k += 32) {
            float hv = h[(long)b * 8192 + k];
            s0 += hv * sel_w[k * 3 + 0];
            s1 += hv * sel_w[k * 3 + 1];
            s2 += hv * sel_w[k * 3 + 2];
        }
        for (int o = 16; o; o >>= 1) {
            s0 += __shfl_xor(s0, o, 32);
            s1 += __shfl_xor(s1, o, 32);
            s2 += __shfl_xor(s2, o, 32);
        }
        if (lane == 0) {
            s0 += sel_b[0]; s1 += sel_b[1]; s2 += sel_b[2];
            float mx = fmaxf(s0, fmaxf(s1, s2));
            float e0 = expf(s0 - mx), e1 = expf(s1 - mx), e2 = expf(s2 - mx);
            float den = e0 + e1 + e2;
            float p0 = e0 / den, p1 = e1 / den, p2 = e2 / den;
            int idx = 0; float bm = p0;
            if (p1 > bm) { bm = p1; idx = 1; }
            if (p2 > bm) { bm = p2; idx = 2; }
            idx_out[b] = idx;
            float z = p0 * sg_w[0] + p1 * sg_w[1] + p2 * sg_w[2] + sg_b[0];
            synL[b] = 1.f / (1.f + expf(-z));
        }
    }
    __syncthreads();
    if (tid == 0) {
        float acc = sgo_b[0];
        for (int b = 0; b < 128; ++b) acc += synL[b] * sgo_w[b];
        float s = 1.f / (1.f + expf(-acc));
        float d = s - sel_loss[0];
        d_out[128001] = d * d;
    }
}

// ---------------------------------------------------------------------------
// mean NLL over log_softmax(logits)
// ---------------------------------------------------------------------------
__global__ __launch_bounds__(256) void loss_kernel(
    const float* __restrict__ logits, const int* __restrict__ target,
    float* __restrict__ out_loss) {
    __shared__ float part[8];
    const int tid = threadIdx.x, wave = tid >> 5, lane = tid & 31;
    float local = 0.f;
    for (int r = wave; r < 128; r += 8) {
        const float* row = logits + (long)r * 1000;
        float mx = -1e30f;
        for (int j = lane; j < 1000; j += 32) mx = fmaxf(mx, row[j]);
        for (int o = 16; o; o >>= 1) mx = fmaxf(mx, __shfl_xor(mx, o, 32));
        float s = 0.f;
        for (int j = lane; j < 1000; j += 32) s += expf(row[j] - mx);
        for (int o = 16; o; o >>= 1) s += __shfl_xor(s, o, 32);
        if (lane == 0) local += row[target[r]] - mx - logf(s);
    }
    if (lane == 0) part[wave] = local;
    __syncthreads();
    if (tid == 0) {
        float t = 0.f;
        for (int w = 0; w < 8; ++w) t += part[w];
        out_loss[0] = -t / 128.f;
    }
}

// ---------------------------------------------------------------------------
extern "C" void kernel_launch(void* const* d_in, const int* in_sizes, int n_in,
                              void* d_out, int out_size, void* d_ws, size_t ws_size,
                              hipStream_t stream) {
    const float* x      = (const float*)d_in[0];
    const int*   target = (const int*)  d_in[1];
    const float* sloss  = (const float*)d_in[2];
    const float* fc0_w  = (const float*)d_in[3];
    const float* fc0_b  = (const float*)d_in[4];
    const float* sel_w  = (const float*)d_in[5];
    const float* sel_b  = (const float*)d_in[6];
    const float* sg_w   = (const float*)d_in[7];
    const float* sg_b   = (const float*)d_in[8];
    const float* sgo_w  = (const float*)d_in[9];
    const float* sgo_b  = (const float*)d_in[10];
    const float* w1     = (const float*)d_in[11];
    const float* b1     = (const float*)d_in[12];
    const float* w2     = (const float*)d_in[13];
    const float* b2     = (const float*)d_in[14];
    const float* fco_w  = (const float*)d_in[15];
    const float* fco_b  = (const float*)d_in[16];
    float* out = (float*)d_out;

    char* ws = (char*)d_ws;
    float*          h32 = (float*)(ws + 0);                  // 128*8192*4 = 4MB
    unsigned short* h16 = (unsigned short*)(ws + 4194304);   // 2MB
    unsigned short* e1b = (unsigned short*)(ws + 6291456);   // 3*128*4096*2 = 3MB
    unsigned short* e2b = (unsigned short*)(ws + 9437184);   // 3*128*2048*2 = 1.5MB
    int*            idx = (int*)(ws + 11010048);             // 512B

    // fc0: h = relu(x @ fc0_w + b), write fp32 (selector) + bf16 (experts)
    GemmP p{};
    p.Bw = fc0_w; p.bias = fc0_b; p.A32 = x;
    p.out32 = h32; p.out16 = h16;
    p.Kreal = 784; p.lda = 784; p.N = 8192;
    gemm_wmma<true><<<dim3(64, 1), 256, 0, stream>>>(p);

    selector_kernel<<<1, 1024, 0, stream>>>(h32, sel_w, sel_b, sg_w, sg_b,
                                            sgo_w, sgo_b, sloss, idx, out);

    // e1 = relu(h @ w1[e] + b1[e]) for all 3 experts
    p = GemmP{};
    p.Bw = w1; p.bias = b1; p.A16 = h16; p.out16 = e1b;
    p.Kreal = 8192; p.lda = 8192; p.N = 4096;
    p.strideB_e = (long)8192 * 4096; p.strideBias_e = 4096;
    p.strideA_e = 0; p.strideO_e = (long)128 * 4096;
    gemm_wmma<false><<<dim3(32, 3), 256, 0, stream>>>(p);

    // e2 = relu(e1 @ w2[e] + b2[e])
    p = GemmP{};
    p.Bw = w2; p.bias = b2; p.A16 = e1b; p.out16 = e2b;
    p.Kreal = 4096; p.lda = 4096; p.N = 2048;
    p.strideB_e = (long)4096 * 2048; p.strideBias_e = 2048;
    p.strideA_e = (long)128 * 4096; p.strideO_e = (long)128 * 2048;
    gemm_wmma<false><<<dim3(16, 3), 256, 0, stream>>>(p);

    // logits = relu(gather(e2, idx) @ fco_w + fco_b)  -> d_out[0:128000]
    p = GemmP{};
    p.Bw = fco_w; p.bias = fco_b; p.A16 = e2b;
    p.gather = idx; p.gstride = (long)128 * 2048;
    p.out32 = out;
    p.Kreal = 2048; p.lda = 2048; p.N = 1000;
    gemm_wmma<false><<<dim3(8, 1), 256, 0, stream>>>(p);

    loss_kernel<<<1, 256, 0, stream>>>(out, target, out + 128000);
}
